// RelationalMultiheadAttention_20572893348635
// MI455X (gfx1250) — matrix operand order repaired and checked by reading, not compiled, not run on hardware
//
#include <hip/hip_runtime.h>
#include <hip/hip_bf16.h>

#define NTOK 4096
#define EDIM 1024
#define HEADS 16
#define DHEAD 64

typedef __attribute__((ext_vector_type(16))) _Float16 v16h;
typedef __attribute__((ext_vector_type(8)))  _Float16 v8h;
typedef __attribute__((ext_vector_type(8)))  float    v8f;

union V16 { v16h v; v8h h[2]; };

__device__ __forceinline__ v8f wmma_f16(v16h a, v16h b, v8f c) {
  return __builtin_amdgcn_wmma_f32_16x16x32_f16(false, a, false, b, (short)0, c, false, false);
}

// ---------------- fp32 -> f16 convert ----------------
__global__ void cvt_f32_f16(const float* __restrict__ src, _Float16* __restrict__ dst, int n) {
  int i = blockIdx.x * blockDim.x + threadIdx.x;
  int stride = gridDim.x * blockDim.x;
  for (; i < n; i += stride) dst[i] = (_Float16)src[i];
}

// ---- shared 32x64-per-wave GEMM core: acc[2][4] = A(32xE) @ B(Ex64) ----
// arow0/arow1: per-lane A row pointers (rows mt*32+l15 and +16)
// brow0: per-lane B row pointer for column tile 0 (col = nt*64 + l15); tiles step 16*EDIM
__device__ __forceinline__ void gemm_core_32x64(
    const _Float16* __restrict__ arow0, const _Float16* __restrict__ arow1,
    const _Float16* __restrict__ brow0, int kA, int kB, v8f acc[2][4])
{
  for (int kb = 0; kb < EDIM; kb += 32) {
    V16 a0, a1;
    a0.h[0] = *(const v8h*)(arow0 + kb + kB);
    a0.h[1] = *(const v8h*)(arow0 + kb + 8 + kB);
    a1.h[0] = *(const v8h*)(arow1 + kb + kB);
    a1.h[1] = *(const v8h*)(arow1 + kb + 8 + kB);
#pragma unroll
    for (int t = 0; t < 4; ++t) {
      V16 b;
      b.v = *(const v16h*)(brow0 + (size_t)(16 * t) * EDIM + kb + kB);
      acc[0][t] = wmma_f16(a0.v, b.v, acc[0][t]);
      acc[1][t] = wmma_f16(a1.v, b.v, acc[1][t]);
    }
  }
}

// ---------------- QKV projection: C = x @ W.T + b, scattered to head layout ----------------
// dst layout: transposed==0 -> [H][N][D] (for Q,K); transposed==1 -> [H][D][N] (for V)
__global__ __launch_bounds__(256) void qkv_proj(
    const _Float16* __restrict__ xh, const _Float16* __restrict__ wh,
    const float* __restrict__ bias, _Float16* __restrict__ dst, int transposed)
{
  int lane = threadIdx.x & 31;
  int wid  = threadIdx.x >> 5;
  int wg   = blockIdx.x * 8 + wid;          // global 32x64 tile id (2048 total)
  int mt   = wg >> 4;                       // token tile (N/32 = 128)
  int nt   = wg & 15;                       // feature tile (E/64 = 16)
  int half = lane >> 4;
  int l15  = lane & 15;
  int kA   = half * 8;
  int kB   = half * 16;

  const _Float16* arow0 = xh + (size_t)(mt * 32 + l15) * EDIM;
  const _Float16* arow1 = arow0 + (size_t)16 * EDIM;
  int col0 = nt * 64 + l15;
  const _Float16* brow0 = wh + (size_t)col0 * EDIM;

  v8f acc[2][4] = {};
  gemm_core_32x64(arow0, arow1, brow0, kA, kB, acc);

#pragma unroll
  for (int mg = 0; mg < 2; ++mg) {
    int rowbase = mt * 32 + mg * 16 + half * 8;
#pragma unroll
    for (int t = 0; t < 4; ++t) {
      int col = col0 + 16 * t;
      float bv = bias[col];
      int h = col >> 6, d = col & 63;
      if (!transposed) {
        _Float16* out = dst + ((size_t)h * NTOK + rowbase) * DHEAD + d;
#pragma unroll
        for (int i = 0; i < 8; ++i) out[(size_t)i * DHEAD] = (_Float16)(acc[mg][t][i] + bv);
      } else {
        v8h pk;
#pragma unroll
        for (int i = 0; i < 8; ++i) pk[i] = (_Float16)(acc[mg][t][i] + bv);
        *(v8h*)(dst + ((size_t)h * DHEAD + d) * NTOK + rowbase) = pk;
      }
    }
  }
}

// ---------------- Flash attention: 1 block = 16 queries, 1 wave = 1 head ----------------
__global__ __launch_bounds__(512) void attn_fwd(
    const _Float16* __restrict__ qh, const _Float16* __restrict__ kh,
    const _Float16* __restrict__ vt, const float* __restrict__ r,
    const int* __restrict__ mask, _Float16* __restrict__ oh)
{
  __shared__ float    sbias[16][33];          // biased r tile, shared by all 16 heads
  __shared__ _Float16 sp[HEADS][16][40];      // per-wave P scratch (16B aligned rows, conflict-free)

  int lane = threadIdx.x & 31;
  int h    = threadIdx.x >> 5;                // wave id == head
  int half = lane >> 4;
  int l15  = lane & 15;
  int kA   = half * 16;
  int qbase = blockIdx.x * 16;
  const float scale = 0.125f;                 // D^-0.5

  // Q block (16x64) kept in registers in A layout, two 16x32 chunks
  const _Float16* qrow = qh + ((size_t)h * NTOK + qbase + l15) * DHEAD;
  V16 aq[2];
#pragma unroll
  for (int c = 0; c < 2; ++c) {
    aq[c].h[0] = *(const v8h*)(qrow + 32 * c + kA);
    aq[c].h[1] = *(const v8h*)(qrow + 32 * c + 8 + kA);
  }

  float mrow[8], lrow[8];
  v8f o0 = {}, o1 = {}, o2 = {}, o3 = {};
#pragma unroll
  for (int i = 0; i < 8; ++i) { mrow[i] = -1e30f; lrow[i] = 0.0f; }

  const _Float16* kbasep = kh + (size_t)h * NTOK * DHEAD;
  const _Float16* vbase  = vt + (size_t)h * DHEAD * NTOK;
  int kBd = half * 16;
  int rowoff = half * 8;

  for (int kb = 0; kb < NTOK; kb += 32) {
    __syncthreads();                           // protect previous-iteration sbias reads
    {
      int rr = threadIdx.x >> 5, cc = threadIdx.x & 31;   // 512 threads = 16x32 tile
      size_t idx = (size_t)(qbase + rr) * NTOK + kb + cc;
      float rv = r[idx];
      int   mv = mask[idx];
      sbias[rr][cc] = mv ? rv : -1e30f;
      if (kb + 32 < NTOK) {
        __builtin_prefetch(r + idx + 32, 0, 1);
        __builtin_prefetch(mask + idx + 32, 0, 1);
      }
    }
    __syncthreads();

    // S = Q @ K^T for 16x32 key slab: two 16x16 C tiles, K-accumulated over D=64
    v8f s0 = {}, s1 = {};
    {
      const _Float16* krow = kbasep + (size_t)(kb + l15) * DHEAD;
      s0 = wmma_f16(aq[0].v, *(const v16h*)(krow + kBd), s0);
      s0 = wmma_f16(aq[1].v, *(const v16h*)(krow + 32 + kBd), s0);
      krow = kbasep + (size_t)(kb + 16 + l15) * DHEAD;
      s1 = wmma_f16(aq[0].v, *(const v16h*)(krow + kBd), s1);
      s1 = wmma_f16(aq[1].v, *(const v16h*)(krow + 32 + kBd), s1);
    }

    // online softmax (rows split: lanes 0-15 rows 0-7, lanes 16-31 rows 8-15)
#pragma unroll
    for (int i = 0; i < 8; ++i) {
      float sv0 = s0[i] * scale + sbias[rowoff + i][l15];
      float sv1 = s1[i] * scale + sbias[rowoff + i][16 + l15];
      float t = fmaxf(sv0, sv1);
#pragma unroll
      for (int off = 1; off < 16; off <<= 1) t = fmaxf(t, __shfl_xor(t, off, 32));
      float mnew  = fmaxf(mrow[i], t);
      float alpha = __expf(mrow[i] - mnew);
      mrow[i] = mnew;
      float p0 = __expf(sv0 - mnew);
      float p1 = __expf(sv1 - mnew);
      float psum = p0 + p1;
#pragma unroll
      for (int off = 1; off < 16; off <<= 1) psum += __shfl_xor(psum, off, 32);
      lrow[i] = lrow[i] * alpha + psum;
      o0[i] *= alpha; o1[i] *= alpha; o2[i] *= alpha; o3[i] *= alpha;
      sp[h][rowoff + i][l15]      = (_Float16)p0;
      sp[h][rowoff + i][16 + l15] = (_Float16)p1;
    }
    asm volatile("" ::: "memory");   // per-wave LDS round-trip; DS ops stay in order in-wave

    // reload P in A layout, then O += P @ V (4 d-tiles of 16)
    V16 ap;
    ap.h[0] = *(const v8h*)(&sp[h][l15][kA]);
    ap.h[1] = *(const v8h*)(&sp[h][l15][8 + kA]);
    int kBv = half * 16;
    o0 = wmma_f16(ap.v, *(const v16h*)(vbase + (size_t)(0 * 16 + l15) * NTOK + kb + kBv), o0);
    o1 = wmma_f16(ap.v, *(const v16h*)(vbase + (size_t)(1 * 16 + l15) * NTOK + kb + kBv), o1);
    o2 = wmma_f16(ap.v, *(const v16h*)(vbase + (size_t)(2 * 16 + l15) * NTOK + kb + kBv), o2);
    o3 = wmma_f16(ap.v, *(const v16h*)(vbase + (size_t)(3 * 16 + l15) * NTOK + kb + kBv), o3);
  }

  // epilogue: normalize and scatter to [N][E] f16
#pragma unroll
  for (int i = 0; i < 8; ++i) {
    float inv = 1.0f / lrow[i];
    _Float16* orow = oh + (size_t)(qbase + rowoff + i) * EDIM + h * DHEAD;
    orow[0 * 16 + l15] = (_Float16)(o0[i] * inv);
    orow[1 * 16 + l15] = (_Float16)(o1[i] * inv);
    orow[2 * 16 + l15] = (_Float16)(o2[i] * inv);
    orow[3 * 16 + l15] = (_Float16)(o3[i] * inv);
  }
}

// ---------------- output projection: out = O @ Wo.T + bo (fp32) ----------------
__global__ __launch_bounds__(256) void out_proj(
    const _Float16* __restrict__ ohm, const _Float16* __restrict__ wh,
    const float* __restrict__ bias, float* __restrict__ out)
{
  int lane = threadIdx.x & 31;
  int wid  = threadIdx.x >> 5;
  int wg   = blockIdx.x * 8 + wid;
  int mt   = wg >> 4;                       // N/32 = 128
  int nt   = wg & 15;                       // E/64 = 16
  int half = lane >> 4;
  int l15  = lane & 15;
  int kA   = half * 8;
  int kB   = half * 16;

  const _Float16* arow0 = ohm + (size_t)(mt * 32 + l15) * EDIM;
  const _Float16* arow1 = arow0 + (size_t)16 * EDIM;
  int col0 = nt * 64 + l15;
  const _Float16* brow0 = wh + (size_t)col0 * EDIM;

  v8f acc[2][4] = {};
  gemm_core_32x64(arow0, arow1, brow0, kA, kB, acc);

#pragma unroll
  for (int mg = 0; mg < 2; ++mg) {
    int rowbase = mt * 32 + mg * 16 + half * 8;
#pragma unroll
    for (int t = 0; t < 4; ++t) {
      int col = col0 + 16 * t;
      float bv = bias[col];
      float* o = out + (size_t)rowbase * EDIM + col;
#pragma unroll
      for (int i = 0; i < 8; ++i) o[(size_t)i * EDIM] = acc[mg][t][i] + bv;
    }
  }
}

extern "C" void kernel_launch(void* const* d_in, const int* in_sizes, int n_in,
                              void* d_out, int out_size, void* d_ws, size_t ws_size,
                              hipStream_t stream) {
  (void)in_sizes; (void)n_in; (void)out_size; (void)ws_size;
  const float* x    = (const float*)d_in[0];
  const float* r    = (const float*)d_in[1];
  const int*   mask = (const int*)d_in[2];
  const float* Wq   = (const float*)d_in[3];
  const float* bq   = (const float*)d_in[4];
  const float* Wk   = (const float*)d_in[5];
  const float* bk   = (const float*)d_in[6];
  const float* Wv   = (const float*)d_in[7];
  const float* bv   = (const float*)d_in[8];
  const float* Wo   = (const float*)d_in[9];
  const float* bo   = (const float*)d_in[10];

  char* ws = (char*)d_ws;
  size_t off = 0;
  auto take = [&](size_t bytes) -> char* {
    char* p = ws + off;
    off += (bytes + 255) & ~(size_t)255;
    return p;
  };
  const size_t NE = (size_t)NTOK * EDIM;
  const size_t EE = (size_t)EDIM * EDIM;
  _Float16* xh  = (_Float16*)take(NE * 2);
  _Float16* wqh = (_Float16*)take(EE * 2);
  _Float16* wkh = (_Float16*)take(EE * 2);
  _Float16* wvh = (_Float16*)take(EE * 2);
  _Float16* woh = (_Float16*)take(EE * 2);
  _Float16* qbuf = (_Float16*)take(NE * 2);   // [H][N][D]
  _Float16* kbuf = (_Float16*)take(NE * 2);   // [H][N][D]
  _Float16* vtb  = (_Float16*)take(NE * 2);   // [H][D][N]
  _Float16* obuf = (_Float16*)take(NE * 2);   // [N][E]

  cvt_f32_f16<<<4096, 256, 0, stream>>>(x,  xh,  (int)NE);
  cvt_f32_f16<<<1024, 256, 0, stream>>>(Wq, wqh, (int)EE);
  cvt_f32_f16<<<1024, 256, 0, stream>>>(Wk, wkh, (int)EE);
  cvt_f32_f16<<<1024, 256, 0, stream>>>(Wv, wvh, (int)EE);
  cvt_f32_f16<<<1024, 256, 0, stream>>>(Wo, woh, (int)EE);

  qkv_proj<<<256, 256, 0, stream>>>(xh, wqh, bq, qbuf, 0);
  qkv_proj<<<256, 256, 0, stream>>>(xh, wkh, bk, kbuf, 0);
  qkv_proj<<<256, 256, 0, stream>>>(xh, wvh, bv, vtb, 1);

  attn_fwd<<<NTOK / 16, 512, 0, stream>>>(qbuf, kbuf, vtb, r, mask, obuf);

  out_proj<<<256, 256, 0, stream>>>(obuf, woh, bo, (float*)d_out);
}
